// GGCN_10565619548474
// MI455X (gfx1250) — compile-verified, matrix operand order checked by reading, not run on hardware
//
#include <hip/hip_runtime.h>

// ---------------------------------------------------------------------------
// GCN forward for MI455X (gfx1250, wave32, WMMA):
//   g1 = relu(a @ (x@W1) + b1); g2 = relu(a @ (g1@W2) + b2);
//   out = relu(g2@Wd1 + bd1) @ Wd2 + bd2
// The two a@X products dominate (34.4 GFLOP vs 1 GB of `a` each) -> stream
// `a` f32 from HBM non-temporally, convert to bf16 in registers, and run
// v_wmma_f32_16x16x32_bf16 at the ~46us/pass HBM roofline.
// ---------------------------------------------------------------------------

typedef __attribute__((ext_vector_type(16))) __bf16          v16bf;
typedef __attribute__((ext_vector_type(8)))  float           v8f;
typedef __attribute__((ext_vector_type(4)))  float           v4f;
typedef __attribute__((ext_vector_type(8)))  unsigned short  v8us;

static constexpr int NN = 16384;  // nodes (M and K of a@X)
static constexpr int CC = 64;     // feature columns of X/Y

__device__ __forceinline__ float bf2f(unsigned short u) {
  return __uint_as_float(((unsigned int)u) << 16);
}
__device__ __forceinline__ unsigned short f2bf(float f) {
  // round-to-nearest-even f32 -> bf16
  unsigned int u = __float_as_uint(f);
  unsigned int r = (u + 0x7FFFu + ((u >> 16) & 1u)) >> 16;
  return (unsigned short)r;
}

// Build one 16x32 bf16 A-fragment for this lane from 16 streamed f32 values:
// two contiguous 8-element K runs at p[0..7] and p[16..23] (ISA A layout).
__device__ __forceinline__ v16bf a_frag_nt(const float* p) {
  const v4f* p4 = (const v4f*)p;
  v4f f0 = __builtin_nontemporal_load(p4 + 0);
  v4f f1 = __builtin_nontemporal_load(p4 + 1);
  v4f f2 = __builtin_nontemporal_load(p4 + 4);
  v4f f3 = __builtin_nontemporal_load(p4 + 5);
  v16bf av;
  av[0]  = (__bf16)f0[0]; av[1]  = (__bf16)f0[1];
  av[2]  = (__bf16)f0[2]; av[3]  = (__bf16)f0[3];
  av[4]  = (__bf16)f1[0]; av[5]  = (__bf16)f1[1];
  av[6]  = (__bf16)f1[2]; av[7]  = (__bf16)f1[3];
  av[8]  = (__bf16)f2[0]; av[9]  = (__bf16)f2[1];
  av[10] = (__bf16)f2[2]; av[11] = (__bf16)f2[3];
  av[12] = (__bf16)f3[0]; av[13] = (__bf16)f3[1];
  av[14] = (__bf16)f3[2]; av[15] = (__bf16)f3[3];
  return av;
}

// ---------------------------------------------------------------------------
// Big kernel: Y[N,64] = bf16( relu( A[N,N](f32) @ Xb[N,64](bf16) + bias ) )
// 256 threads = 8 waves; block computes 64 rows x 64 cols; wave -> 16x32.
// K-step 64 (4 WMMAs), double-buffered LDS, one barrier per step.
// ---------------------------------------------------------------------------
__global__ __launch_bounds__(256)
void k_agemm_bias_relu(const float* __restrict__ A,
                       const unsigned short* __restrict__ Xb,
                       const float* __restrict__ bias,
                       unsigned short* __restrict__ Y)
{
  // X chunk (64 k x 64 cols) transposed: Xt[buf][col][k], row stride 72
  // shorts (144B) keeps 16B ds_load_b128 reads conflict-free across banks.
  __shared__ __attribute__((aligned(16))) unsigned short Xt[2][64][72];

  const int tid  = threadIdx.x;
  const int lane = tid & 31;
  const int wave = tid >> 5;
  const int wr   = wave >> 1;        // 0..3 : 16-row slice
  const int wc   = wave & 1;         // 0..1 : 32-col half
  const int ln16 = lane & 15;
  const int kh   = lane >> 4;        // which K half-group this lane holds

  const int rowA = blockIdx.x * 64 + wr * 16 + ln16;  // A row this lane loads
  const int c0   = wc * 32 + ln16;                    // first output column
  const int c1   = c0 + 16;                           // second output column

  // staging mapping: thread t loads 8 contiguous bf16 (one b128) per half
  const int skk = tid >> 3;          // k within chunk, 0..31 (+32 second half)
  const int sc  = (tid & 7) * 8;     // starting column, 0..56

  v8f acc0 = {};
  v8f acc1 = {};

  const float* arow = A + (size_t)rowA * NN + kh * 8;

  // ---- prologue: stage chunk k=[0,64) into buffer 0 ----
#pragma unroll
  for (int i = 0; i < 2; ++i) {
    int kk = skk + i * 32;
    v8us d = *(const v8us*)(Xb + (size_t)kk * CC + sc);
#pragma unroll
    for (int j = 0; j < 8; ++j) Xt[0][sc + j][kk] = d[j];
  }
  __syncthreads();

  for (int k0 = 0; k0 < NN; k0 += 64) {
    const int buf = (k0 >> 6) & 1;
    const bool haveNext = (k0 + 64) < NN;

    // -- start next chunk's global loads early (consumed after the WMMAs) --
    v8us st0 = {}, st1 = {};
    if (haveNext) {
      st0 = *(const v8us*)(Xb + (size_t)(k0 + 64 + skk) * CC + sc);
      st1 = *(const v8us*)(Xb + (size_t)(k0 + 96 + skk) * CC + sc);
    }

    // -- B fragments (column-major out of transposed LDS), issued first so
    //    the dscnt wait for them doesn't cover this iteration's ds stores --
    union BU { uint4 q[2]; v16bf v; };
    BU b00, b01, b10, b11;
    b00.q[0] = *(const uint4*)&Xt[buf][c0][kh * 8];
    b00.q[1] = *(const uint4*)&Xt[buf][c0][16 + kh * 8];
    b01.q[0] = *(const uint4*)&Xt[buf][c1][kh * 8];
    b01.q[1] = *(const uint4*)&Xt[buf][c1][16 + kh * 8];
    b10.q[0] = *(const uint4*)&Xt[buf][c0][32 + kh * 8];
    b10.q[1] = *(const uint4*)&Xt[buf][c0][48 + kh * 8];
    b11.q[0] = *(const uint4*)&Xt[buf][c1][32 + kh * 8];
    b11.q[1] = *(const uint4*)&Xt[buf][c1][48 + kh * 8];

    // -- A fragments: stream f32 non-temporally, cvt_pk to bf16 --
    v16bf av0 = a_frag_nt(arow + k0);
    v16bf av1 = a_frag_nt(arow + k0 + 32);

    acc0 = __builtin_amdgcn_wmma_f32_16x16x32_bf16(
        false, av0, false, b00.v, (short)0, acc0, false, false);
    acc1 = __builtin_amdgcn_wmma_f32_16x16x32_bf16(
        false, av0, false, b01.v, (short)0, acc1, false, false);
    acc0 = __builtin_amdgcn_wmma_f32_16x16x32_bf16(
        false, av1, false, b10.v, (short)0, acc0, false, false);
    acc1 = __builtin_amdgcn_wmma_f32_16x16x32_bf16(
        false, av1, false, b11.v, (short)0, acc1, false, false);

    // -- scatter staged chunk into the other buffer (uniform branch) --
    if (haveNext) {
#pragma unroll
      for (int j = 0; j < 8; ++j) Xt[buf ^ 1][sc + j][skk] = st0[j];
#pragma unroll
      for (int j = 0; j < 8; ++j) Xt[buf ^ 1][sc + j][skk + 32] = st1[j];
    }
    __syncthreads();
  }

  // ---- epilogue: C/D layout lane=N, vgpr=M (+8 for lanes 16..31) ----
  const int rowTop = blockIdx.x * 64 + wr * 16 + kh * 8;
  const float bb0 = bias[c0];
  const float bb1 = bias[c1];
#pragma unroll
  for (int g = 0; g < 8; ++g) {
    int r = rowTop + g;
    float v0 = acc0[g] + bb0; v0 = v0 > 0.f ? v0 : 0.f;
    float v1 = acc1[g] + bb1; v1 = v1 > 0.f ? v1 : 0.f;
    Y[(size_t)r * CC + c0] = f2bf(v0);
    Y[(size_t)r * CC + c1] = f2bf(v1);
  }
}

// ---------------------------------------------------------------------------
// XW1b = bf16( x[N,32] @ W1[32,64] )   (one thread per output element)
// ---------------------------------------------------------------------------
__global__ __launch_bounds__(256)
void k_xw1(const float* __restrict__ x, const float* __restrict__ W1,
           unsigned short* __restrict__ out)
{
  int t   = blockIdx.x * 256 + threadIdx.x;  // N*64 threads
  int row = t >> 6;
  int col = t & 63;
  const float* xr = x + (size_t)row * 32;
  float s = 0.f;
#pragma unroll
  for (int k = 0; k < 32; ++k) s += xr[k] * W1[k * 64 + col];
  out[t] = f2bf(s);
}

// ---------------------------------------------------------------------------
// Hb = bf16( g1b[N,64](bf16) @ W[64,64] )
// ---------------------------------------------------------------------------
__global__ __launch_bounds__(256)
void k_gemm64(const unsigned short* __restrict__ Xb, const float* __restrict__ W,
              unsigned short* __restrict__ out)
{
  int t   = blockIdx.x * 256 + threadIdx.x;  // N*64 threads
  int row = t >> 6;
  int col = t & 63;
  const unsigned short* xr = Xb + (size_t)row * 64;
  float s = 0.f;
#pragma unroll
  for (int k = 0; k < 64; ++k) s += bf2f(xr[k]) * W[k * 64 + col];
  out[t] = f2bf(s);
}

// ---------------------------------------------------------------------------
// out[row] = relu(g2[row,:] @ Wd1 + bd1) @ Wd2 + bd2   (one thread per row)
// ---------------------------------------------------------------------------
__global__ __launch_bounds__(256)
void k_head(const unsigned short* __restrict__ g2, const float* __restrict__ Wd1,
            const float* __restrict__ bd1, const float* __restrict__ Wd2,
            const float* __restrict__ bd2, float* __restrict__ out)
{
  int row = blockIdx.x * 256 + threadIdx.x;  // N threads
  float g[64];
  const unsigned short* gr = g2 + (size_t)row * 64;
#pragma unroll
  for (int k = 0; k < 64; ++k) g[k] = bf2f(gr[k]);

  float r = bd2[0];
  for (int c = 0; c < 32; ++c) {
    float s = bd1[c];
#pragma unroll
    for (int k = 0; k < 64; ++k) s += g[k] * Wd1[k * 32 + c];
    s = s > 0.f ? s : 0.f;
    r += s * Wd2[c];
  }
  out[row] = r;
}

// ---------------------------------------------------------------------------
extern "C" void kernel_launch(void* const* d_in, const int* in_sizes, int n_in,
                              void* d_out, int out_size, void* d_ws, size_t ws_size,
                              hipStream_t stream)
{
  (void)in_sizes; (void)n_in; (void)out_size; (void)ws_size;
  const float* x   = (const float*)d_in[0];
  const float* a   = (const float*)d_in[1];
  const float* W1  = (const float*)d_in[2];
  const float* b1  = (const float*)d_in[3];
  const float* W2  = (const float*)d_in[4];
  const float* b2  = (const float*)d_in[5];
  const float* Wd1 = (const float*)d_in[6];
  const float* bd1 = (const float*)d_in[7];
  const float* Wd2 = (const float*)d_in[8];
  const float* bd2 = (const float*)d_in[9];

  // workspace: four bf16 [N,64] buffers (2 MB each)
  unsigned short* XW1b = (unsigned short*)d_ws;
  unsigned short* g1b  = XW1b + (size_t)NN * CC;
  unsigned short* Hb   = g1b  + (size_t)NN * CC;
  unsigned short* g2b  = Hb   + (size_t)NN * CC;

  const int elemBlocks = (NN * CC) / 256;  // 4096

  k_xw1<<<elemBlocks, 256, 0, stream>>>(x, W1, XW1b);
  k_agemm_bias_relu<<<NN / 64, 256, 0, stream>>>(a, XW1b, b1, g1b);
  k_gemm64<<<elemBlocks, 256, 0, stream>>>(g1b, W2, Hb);
  k_agemm_bias_relu<<<NN / 64, 256, 0, stream>>>(a, Hb, b2, g2b);
  k_head<<<NN / 256, 256, 0, stream>>>(g2b, Wd1, bd1, Wd2, bd2, (float*)d_out);
}
